// DualTrackMamba_11106785427543
// MI455X (gfx1250) — matrix operand order repaired and checked by reading, not compiled, not run on hardware
//
#include <hip/hip_runtime.h>
#include <hip/hip_bf16.h>
#include <stddef.h>

// ---------------------------------------------------------------------------
// Dual-track Mamba (B=1, L=2048, d_model=1024, d_inner=2048, n=16, dt_rank=64)
// GEMMs: bf16 v_wmma_f32_16x16x32_bf16, 128x128x32 block tiles, LDS
// double-buffered, 64x32 wave tiles. Selective scan on VALU with LDS staging.
// ---------------------------------------------------------------------------

#define LSEQ    2048
#define DMODEL  1024
#define DINNER  2048
#define DSTATE  16
#define DTRANK  64
#define DCONV   4
#define XDBLW   (DTRANK + 2 * DSTATE)   // 96

#define BK      32
#define LDS_STRIDE 40                    // bf16 elements per LDS row (32 + 8 pad)

typedef __attribute__((ext_vector_type(16))) __bf16 v16bf;
typedef __attribute__((ext_vector_type(8)))  __bf16 v8bf;
typedef __attribute__((ext_vector_type(8)))  float  v8f;
typedef __attribute__((ext_vector_type(4)))  unsigned int v4u;

__device__ __forceinline__ unsigned short f2bf(float f) {
    union { float f; unsigned int u; } v; v.f = f;
    unsigned int r = v.u + 0x7FFFu + ((v.u >> 16) & 1u);   // round to nearest even
    return (unsigned short)(r >> 16);
}

__device__ __forceinline__ float silu(float x) {
    return x / (1.0f + __expf(-x));
}

// Build a 16x bf16 fragment from two 16-byte LDS loads.
__device__ __forceinline__ v16bf ld_frag(const unsigned short* p, int off_lo, int off_hi) {
    v8bf lo = *(const v8bf*)(p + off_lo);
    v8bf hi = *(const v8bf*)(p + off_hi);
    return __builtin_shufflevector(lo, hi,
            0,1,2,3,4,5,6,7,8,9,10,11,12,13,14,15);
}

// --------------------------- f32 -> bf16 convert ---------------------------
__global__ void cvt_f32_bf16(const float* __restrict__ src,
                             unsigned short* __restrict__ dst, int n) {
    int i = blockIdx.x * blockDim.x + threadIdx.x;
    if (i < n) dst[i] = f2bf(src[i]);
}

// --------------------------- WMMA bf16 GEMM --------------------------------
// C[M,N] (f32, row stride ldc) = A[M,K](bf16 row-major) * W[N,K]^T (bf16).
// Block tile 128x128, BK=32, LDS double buffered. 8 waves: 2(M) x 4(N),
// wave tile 64x32 = 4x2 WMMA accumulators.
// grid = ((N+127)/128, M/128), block = 256.
__global__ __launch_bounds__(256)
void gemm_bf16_wmma(const unsigned short* __restrict__ Au,
                    const unsigned short* __restrict__ Wu,
                    float* __restrict__ C,
                    int M, int N, int K, int ldc) {
    __shared__ unsigned short sA[2][128 * LDS_STRIDE];
    __shared__ unsigned short sB[2][128 * LDS_STRIDE];

    const int tid   = threadIdx.x;
    const int lane  = tid & 31;
    const int wave  = tid >> 5;
    const int waveM = wave >> 2;                 // 0..1
    const int waveN = wave & 3;                  // 0..3
    const int half  = lane >> 4;                 // 0: lanes 0-15, 1: lanes 16-31
    const int l16   = lane & 15;

    const int bm0 = blockIdx.y * 128;
    const int bn0 = blockIdx.x * 128;

    // Global staging: 512 16-byte chunks per tile (128 rows x 4 chunks).
    // Thread t handles chunks {t, t+256}: row = t>>2 (+64), col = t&3.
    const int srow = tid >> 2;
    const int scol = tid & 3;
    const unsigned short* gA0 = Au + (size_t)(bm0 + srow) * K + scol * 8;
    const unsigned short* gA1 = Au + (size_t)(bm0 + srow + 64) * K + scol * 8;
    const int brow0 = min(bn0 + srow,      N - 1);   // clamp for ragged N
    const int brow1 = min(bn0 + srow + 64, N - 1);
    const unsigned short* gB0 = Wu + (size_t)brow0 * K + scol * 8;
    const unsigned short* gB1 = Wu + (size_t)brow1 * K + scol * 8;
    const int ldsOff = srow * LDS_STRIDE + scol * 8;         // elements

    const v8f vzero = {0.f,0.f,0.f,0.f,0.f,0.f,0.f,0.f};
    v8f acc[4][2];
    #pragma unroll
    for (int i = 0; i < 4; ++i)
        #pragma unroll
        for (int j = 0; j < 2; ++j)
            acc[i][j] = vzero;

    // ---- prologue: stage K-slice 0 ----
    v4u ra0 = *(const v4u*)(gA0);
    v4u ra1 = *(const v4u*)(gA1);
    v4u rb0 = *(const v4u*)(gB0);
    v4u rb1 = *(const v4u*)(gB1);
    *(v4u*)(&sA[0][ldsOff])                   = ra0;
    *(v4u*)(&sA[0][ldsOff + 64 * LDS_STRIDE]) = ra1;
    *(v4u*)(&sB[0][ldsOff])                   = rb0;
    *(v4u*)(&sB[0][ldsOff + 64 * LDS_STRIDE]) = rb1;
    __syncthreads();

    const int nK = K / BK;
    int buf = 0;
    for (int kb = 0; kb < nK; ++kb) {
        const bool has_next = (kb + 1 < nK);
        if (has_next) {
            const int kn = (kb + 1) * BK;
            if (kb + 2 < nK) {                   // gfx1250 global_prefetch_b8
                __builtin_prefetch(gA0 + kn + BK, 0, 0);
                __builtin_prefetch(gB0 + kn + BK, 0, 0);
            }
            ra0 = *(const v4u*)(gA0 + kn);
            ra1 = *(const v4u*)(gA1 + kn);
            rb0 = *(const v4u*)(gB0 + kn);
            rb1 = *(const v4u*)(gB1 + kn);
        }

        // ---- compute from LDS buffer `buf` ----
        // B fragment (32x16): lane = col (n0+l16); elems 0..7 -> K = 8*half+0..7 ; elems 8..15 -> K = 16+8*half+0..7
        v16bf bfrag[2];
        #pragma unroll
        for (int j = 0; j < 2; ++j) {
            const int brow = waveN * 32 + j * 16 + l16;
            bfrag[j] = ld_frag(&sB[buf][0],
                               brow * LDS_STRIDE + 8 * half,
                               brow * LDS_STRIDE + 16 + 8 * half);
        }
        // A fragment (16x32): lane = row (m0+l16);
        //   elems 0..7 -> K = 8*half+0..7 ; elems 8..15 -> K = 16+8*half+0..7
        #pragma unroll
        for (int i = 0; i < 4; ++i) {
            const int arow = waveM * 64 + i * 16 + l16;
            v16bf afrag = ld_frag(&sA[buf][0],
                                  arow * LDS_STRIDE + 8 * half,
                                  arow * LDS_STRIDE + 16 + 8 * half);
            #pragma unroll
            for (int j = 0; j < 2; ++j)
                acc[i][j] = __builtin_amdgcn_wmma_f32_16x16x32_bf16(
                                false, afrag, false, bfrag[j],
                                (short)0, acc[i][j], false, false);
        }

        if (has_next) {
            const int nb = buf ^ 1;
            *(v4u*)(&sA[nb][ldsOff])                   = ra0;
            *(v4u*)(&sA[nb][ldsOff + 64 * LDS_STRIDE]) = ra1;
            *(v4u*)(&sB[nb][ldsOff])                   = rb0;
            *(v4u*)(&sB[nb][ldsOff + 64 * LDS_STRIDE]) = rb1;
            __syncthreads();
            buf = nb;
        }
    }

    // ---- epilogue: D elem r -> (m = r + 8*half, n = l16) within each tile ----
    #pragma unroll
    for (int i = 0; i < 4; ++i) {
        const int mrow = bm0 + waveM * 64 + i * 16 + half * 8;
        #pragma unroll
        for (int j = 0; j < 2; ++j) {
            const int ncol = bn0 + waveN * 32 + j * 16;
            if (ncol < N) {                       // uniform across the wave
                float* crow = C + (size_t)mrow * ldc + ncol + l16;
                #pragma unroll
                for (int r = 0; r < 8; ++r)
                    crow[(size_t)r * ldc] = acc[i][j][r];
            }
        }
    }
}

// ------------------ causal depthwise conv (width 4) + SiLU -----------------
__global__ void conv_silu_kernel(const float* __restrict__ xz,
                                 const float* __restrict__ cw,
                                 const float* __restrict__ cb,
                                 float* __restrict__ xcf,
                                 unsigned short* __restrict__ xcb) {
    int i = blockIdx.x * blockDim.x + threadIdx.x;
    if (i >= LSEQ * DINNER) return;
    int l = i >> 11;
    int d = i & (DINNER - 1);
    float s = cb[d];
    #pragma unroll
    for (int t = 0; t < DCONV; ++t) {
        int lt = l - (DCONV - 1) + t;
        if (lt >= 0) s += cw[d * DCONV + t] * xz[(size_t)lt * (2 * DINNER) + d];
    }
    float y = silu(s);
    xcf[i] = y;
    xcb[i] = f2bf(y);
}

// ---------------- extract dt_lo (cols 0..63 of xdbl) as bf16 ---------------
__global__ void extract_dtlo_kernel(const float* __restrict__ xdbl,
                                    unsigned short* __restrict__ dst) {
    int i = blockIdx.x * blockDim.x + threadIdx.x;
    if (i >= LSEQ * DTRANK) return;
    int l = i >> 6;
    int c = i & (DTRANK - 1);
    dst[i] = f2bf(xdbl[(size_t)l * XDBLW + c]);
}

// ---------------- dt = softplus(dtproj + dt_b), in place -------------------
__global__ void softplus_bias_kernel(float* __restrict__ dt,
                                     const float* __restrict__ dtb) {
    int i = blockIdx.x * blockDim.x + threadIdx.x;
    if (i >= LSEQ * DINNER) return;
    int d = i & (DINNER - 1);
    float x = dt[i] + dtb[d];
    dt[i] = (x > 20.0f) ? x : log1pf(__expf(x));
}

// --------------------------- selective scan --------------------------------
__global__ void scan_kernel(const float* __restrict__ u,
                            const float* __restrict__ dt,
                            const float* __restrict__ xdbl,
                            const float* __restrict__ Alog,
                            const float* __restrict__ Dp,
                            float* __restrict__ y) {
    int d = blockIdx.x * blockDim.x + threadIdx.x;   // d in [0, DINNER)
    float A[DSTATE], h[DSTATE];
    #pragma unroll
    for (int n = 0; n < DSTATE; ++n) {
        A[n] = -__expf(Alog[(size_t)d * DSTATE + n]);
        h[n] = 0.0f;
    }
    float Dd = Dp[d];
    __shared__ float sBC[2 * DSTATE];

    for (int l = 0; l < LSEQ; ++l) {
        if (threadIdx.x < 2 * DSTATE)
            sBC[threadIdx.x] = xdbl[(size_t)l * XDBLW + DTRANK + threadIdx.x];
        __syncthreads();
        float dtl = dt[(size_t)l * DINNER + d];
        float ul  = u [(size_t)l * DINNER + d];
        float du  = dtl * ul;
        float acc = 0.0f;
        #pragma unroll
        for (int n = 0; n < DSTATE; ++n) {
            float hb = __expf(dtl * A[n]) * h[n] + du * sBC[n];
            h[n] = hb;
            acc += hb * sBC[DSTATE + n];
        }
        y[(size_t)l * DINNER + d] = acc + ul * Dd;
        __syncthreads();
    }
}

// ---------------------- y * silu(z)  -> bf16 -------------------------------
__global__ void gate_kernel(const float* __restrict__ y,
                            const float* __restrict__ xz,
                            unsigned short* __restrict__ yg) {
    int i = blockIdx.x * blockDim.x + threadIdx.x;
    if (i >= LSEQ * DINNER) return;
    int l = i >> 11;
    int d = i & (DINNER - 1);
    float z = xz[(size_t)l * (2 * DINNER) + DINNER + d];
    yg[i] = f2bf(y[i] * silu(z));
}

// ------------- softmax(track_weights) blend  -> bf16 -----------------------
__global__ void combine_kernel(const float* __restrict__ tw,
                               const float* __restrict__ fo,
                               const float* __restrict__ so,
                               unsigned short* __restrict__ comb) {
    int i = blockIdx.x * blockDim.x + threadIdx.x;
    if (i >= LSEQ * DMODEL) return;
    float w0 = tw[0], w1 = tw[1];
    float m  = fmaxf(w0, w1);
    float e0 = __expf(w0 - m), e1 = __expf(w1 - m);
    float a  = e0 / (e0 + e1);
    comb[i] = f2bf(a * fo[i] + (1.0f - a) * so[i]);
}

// --------------------------- out += proj_b ---------------------------------
__global__ void bias_add_kernel(float* __restrict__ out,
                                const float* __restrict__ b) {
    int i = blockIdx.x * blockDim.x + threadIdx.x;
    if (i >= LSEQ * DMODEL) return;
    out[i] += b[i & (DMODEL - 1)];
}

// ===========================================================================
// Host launch
// ===========================================================================
static inline void launch_gemm(const unsigned short* A, const unsigned short* W,
                               float* C, int M, int N, int K, int ldc,
                               hipStream_t s) {
    dim3 grid((N + 127) / 128, M / 128);
    gemm_bf16_wmma<<<grid, dim3(256), 0, s>>>(A, W, C, M, N, K, ldc);
}

extern "C" void kernel_launch(void* const* d_in, const int* in_sizes, int n_in,
                              void* d_out, int out_size, void* d_ws, size_t ws_size,
                              hipStream_t stream) {
    (void)in_sizes; (void)n_in; (void)out_size; (void)ws_size;

    // Input order: hidden, fast{in_w,conv_w,conv_b,xproj_w,dt_w,dt_b,A_log,D,out_w},
    //              slow{...}, track_weights, proj_w, proj_b
    const float* hidden = (const float*)d_in[0];
    const float* tw     = (const float*)d_in[19];
    const float* projw  = (const float*)d_in[20];
    const float* projb  = (const float*)d_in[21];

    char*  ws  = (char*)d_ws;
    size_t off = 0;
    auto alloc = [&](size_t bytes) -> char* {
        char* p = ws + off;
        off += (bytes + 255) & ~(size_t)255;
        return p;
    };

    unsigned short* hb       = (unsigned short*)alloc((size_t)LSEQ * DMODEL * 2);
    unsigned short* projw_bf = (unsigned short*)alloc((size_t)DMODEL * DMODEL * 2);
    unsigned short* inw_bf   = (unsigned short*)alloc((size_t)2 * DINNER * DMODEL * 2);
    unsigned short* xprojw_bf= (unsigned short*)alloc((size_t)XDBLW * DINNER * 2);
    unsigned short* dtw_bf   = (unsigned short*)alloc((size_t)DINNER * DTRANK * 2);
    unsigned short* outw_bf  = (unsigned short*)alloc((size_t)DMODEL * DINNER * 2);
    float*          xz       = (float*)alloc((size_t)LSEQ * 2 * DINNER * 4);
    float*          xconv_f  = (float*)alloc((size_t)LSEQ * DINNER * 4);
    unsigned short* xconv_bf = (unsigned short*)alloc((size_t)LSEQ * DINNER * 2);
    float*          xdbl     = (float*)alloc((size_t)LSEQ * XDBLW * 4);
    unsigned short* dtlo_bf  = (unsigned short*)alloc((size_t)LSEQ * DTRANK * 2);
    float*          dtbuf    = (float*)alloc((size_t)LSEQ * DINNER * 4);
    float*          ybuf     = (float*)alloc((size_t)LSEQ * DINNER * 4);
    unsigned short* yg_bf    = (unsigned short*)alloc((size_t)LSEQ * DINNER * 2);
    float*          fast_out = (float*)alloc((size_t)LSEQ * DMODEL * 4);
    float*          slow_out = (float*)alloc((size_t)LSEQ * DMODEL * 4);
    unsigned short* comb_bf  = (unsigned short*)alloc((size_t)LSEQ * DMODEL * 2);

    const int TB = 256;
    auto nblk = [](int n) { return (n + 255) / 256; };

    // Shared conversions
    cvt_f32_bf16<<<nblk(LSEQ * DMODEL), TB, 0, stream>>>(hidden, hb, LSEQ * DMODEL);
    cvt_f32_bf16<<<nblk(DMODEL * DMODEL), TB, 0, stream>>>(projw, projw_bf, DMODEL * DMODEL);

    for (int t = 0; t < 2; ++t) {
        int base = 1 + t * 9;
        const float* in_w   = (const float*)d_in[base + 0];
        const float* conv_w = (const float*)d_in[base + 1];
        const float* conv_b = (const float*)d_in[base + 2];
        const float* xproj_w= (const float*)d_in[base + 3];
        const float* dt_w   = (const float*)d_in[base + 4];
        const float* dt_b   = (const float*)d_in[base + 5];
        const float* A_log  = (const float*)d_in[base + 6];
        const float* Dp     = (const float*)d_in[base + 7];
        const float* out_w  = (const float*)d_in[base + 8];
        float* trk_out = (t == 0) ? fast_out : slow_out;

        cvt_f32_bf16<<<nblk(2 * DINNER * DMODEL), TB, 0, stream>>>(in_w, inw_bf, 2 * DINNER * DMODEL);
        cvt_f32_bf16<<<nblk(XDBLW * DINNER), TB, 0, stream>>>(xproj_w, xprojw_bf, XDBLW * DINNER);
        cvt_f32_bf16<<<nblk(DINNER * DTRANK), TB, 0, stream>>>(dt_w, dtw_bf, DINNER * DTRANK);
        cvt_f32_bf16<<<nblk(DMODEL * DINNER), TB, 0, stream>>>(out_w, outw_bf, DMODEL * DINNER);

        // xz = h @ in_w^T            [2048 x 4096]
        launch_gemm(hb, inw_bf, xz, LSEQ, 2 * DINNER, DMODEL, 2 * DINNER, stream);
        // x = silu(causal_conv(x))
        conv_silu_kernel<<<nblk(LSEQ * DINNER), TB, 0, stream>>>(xz, conv_w, conv_b, xconv_f, xconv_bf);
        // x_dbl = x @ xproj_w^T      [2048 x 96]
        launch_gemm(xconv_bf, xprojw_bf, xdbl, LSEQ, XDBLW, DINNER, XDBLW, stream);
        // dt = softplus(dt_lo @ dt_w^T + dt_b)
        extract_dtlo_kernel<<<nblk(LSEQ * DTRANK), TB, 0, stream>>>(xdbl, dtlo_bf);
        launch_gemm(dtlo_bf, dtw_bf, dtbuf, LSEQ, DINNER, DTRANK, DINNER, stream);
        softplus_bias_kernel<<<nblk(LSEQ * DINNER), TB, 0, stream>>>(dtbuf, dt_b);
        // selective scan
        scan_kernel<<<dim3(DINNER / TB), TB, 0, stream>>>(xconv_f, dtbuf, xdbl, A_log, Dp, ybuf);
        // y * silu(z) -> bf16
        gate_kernel<<<nblk(LSEQ * DINNER), TB, 0, stream>>>(ybuf, xz, yg_bf);
        // track_out = yg @ out_w^T   [2048 x 1024]
        launch_gemm(yg_bf, outw_bf, trk_out, LSEQ, DMODEL, DINNER, DMODEL, stream);
    }

    // combined = softmax(tw) blend -> bf16
    combine_kernel<<<nblk(LSEQ * DMODEL), TB, 0, stream>>>(tw, fast_out, slow_out, comb_bf);
    // out = combined @ proj_w^T + proj_b
    launch_gemm(comb_bf, projw_bf, (float*)d_out, LSEQ, DMODEL, DMODEL, DMODEL, stream);
    bias_add_kernel<<<nblk(LSEQ * DMODEL), TB, 0, stream>>>((float*)d_out, projb);
}
